// SelfAttention_59021440582207
// MI455X (gfx1250) — matrix operand order repaired and checked by reading, not compiled, not run on hardware
//
#include <hip/hip_runtime.h>
#include <stdint.h>

// ---------------------------------------------------------------------------
// Problem constants (B,N,D)=(2,2048,2048), H=32, HD=64, KVH=8, G=4
// ---------------------------------------------------------------------------
constexpr int Bb   = 2;
constexpr int Nn   = 2048;
constexpr int Dd   = 2048;
constexpr int Hh   = 32;
constexpr int HDd  = 64;
constexpr int KVHh = 8;
constexpr int Mrows = Bb * Nn;          // 4096

typedef __attribute__((ext_vector_type(16))) __bf16 v16bf;
typedef __attribute__((ext_vector_type(8)))  float  v8f;

union ABfrag { v16bf v; uint4 u[2]; };

__device__ __forceinline__ unsigned short f2bf(float f) {
    union { float f; unsigned int u; } x; x.f = f;
    unsigned int u = x.u;
    unsigned int r = (u + 0x7FFFu + ((u >> 16) & 1u)) >> 16;   // RNE
    return (unsigned short)r;
}
__device__ __forceinline__ float bf2f(unsigned short h) {
    union { unsigned int u; float f; } x; x.u = ((unsigned int)h) << 16;
    return x.f;
}
__device__ __forceinline__ v8f wmma_bf(v16bf a, v16bf b, v8f c) {
    return __builtin_amdgcn_wmma_f32_16x16x32_bf16(false, a, false, b, (short)0, c,
                                                   false, false);
}

// ---------------------------------------------------------------------------
// Vectorized f32 -> bf16 convert (float4 in, 4 packed bf16 out)
// ---------------------------------------------------------------------------
__global__ void k_f32_to_bf16_v4(const float* __restrict__ src,
                                 unsigned short* __restrict__ dst, int n4) {
    int i = blockIdx.x * blockDim.x + threadIdx.x;
    if (i >= n4) return;
    float4 f = ((const float4*)src)[i];
    union { unsigned short s[4]; uint2 u; } o;
    o.s[0] = f2bf(f.x); o.s[1] = f2bf(f.y); o.s[2] = f2bf(f.z); o.s[3] = f2bf(f.w);
    ((uint2*)dst)[i] = o.u;
}

// ---------------------------------------------------------------------------
// LDS-tiled transpose+convert: W[K][Nc] f32 -> Wt[Nc][K] bf16
// 32x32 tile, 256 threads (32x8, 4 rows each), coalesced read AND write.
// ---------------------------------------------------------------------------
__global__ __launch_bounds__(256) void k_wtrans(const float* __restrict__ src,
                                                unsigned short* __restrict__ dst,
                                                int K, int Nc) {
    __shared__ float tile[32][33];
    int nT = Nc >> 5;
    int tk = blockIdx.x / nT, tn = blockIdx.x % nT;
    int k0 = tk << 5, n0 = tn << 5;
    int tx = threadIdx.x & 31, ty = threadIdx.x >> 5;
#pragma unroll
    for (int i = 0; i < 4; i++) {
        int k = ty * 4 + i;
        tile[k][tx] = src[(size_t)(k0 + k) * Nc + n0 + tx];
    }
    __syncthreads();
#pragma unroll
    for (int i = 0; i < 4; i++) {
        int n = ty * 4 + i;
        dst[(size_t)(n0 + n) * K + k0 + tx] = f2bf(tile[tx][n]);
    }
}

// ---------------------------------------------------------------------------
// LDS-tiled V transpose: Vg[B][N][KVH*64] bf16 -> Vt[B][KVH][64][N] bf16
// 32x32 tiles per (b, kv); coalesced read and write.
// ---------------------------------------------------------------------------
__global__ __launch_bounds__(256) void k_vtrans(const unsigned short* __restrict__ src,
                                                unsigned short* __restrict__ dst) {
    __shared__ unsigned short tile[32][34];
    int idx = blockIdx.x;
    int ntile = idx & 63; idx >>= 6;      // N/32 = 64
    int hdt   = idx & 1;  idx >>= 1;      // 64/32 = 2
    int kv    = idx & 7;  idx >>= 3;      // KVH = 8
    int b     = idx;
    int n0 = ntile << 5, hd0 = hdt << 5;
    int tx = threadIdx.x & 31, ty = threadIdx.x >> 5;
#pragma unroll
    for (int i = 0; i < 4; i++) {
        int n = ty * 4 + i;
        tile[n][tx] =
            src[(size_t)(b * Nn + n0 + n) * (KVHh * HDd) + kv * 64 + hd0 + tx];
    }
    __syncthreads();
#pragma unroll
    for (int i = 0; i < 4; i++) {
        int hd = ty * 4 + i;
        dst[((size_t)(b * KVHh + kv) * 64 + hd0 + hd) * Nn + n0 + tx] = tile[tx][hd];
    }
}

// ---------------------------------------------------------------------------
// bf16 WMMA GEMM:  C[M][Nc] = A[M][K] @ Bt[Nc][K]^T
// block = 128x128 tile, 8 waves (2 M x 4 N), wave = 64x32 (4x2 fragments)
// ---------------------------------------------------------------------------
template <bool BF16OUT>
__global__ __launch_bounds__(256) void k_gemm(const unsigned short* __restrict__ A,
                                              const unsigned short* __restrict__ Bt,
                                              void* __restrict__ Cout,
                                              int M, int K, int Nc) {
    const int nTiles = Nc >> 7;
    int bm = blockIdx.x / nTiles, bn = blockIdx.x % nTiles;
    int w = threadIdx.x >> 5, lane = threadIdx.x & 31;
    int lr = lane & 15, lh = lane >> 4;
    int wm = w & 1, wn = w >> 1;
    int m0 = bm * 128 + wm * 64;
    int n0 = bn * 128 + wn * 32;

    v8f acc[4][2];
    v8f zero = {0.f, 0.f, 0.f, 0.f, 0.f, 0.f, 0.f, 0.f};
#pragma unroll
    for (int a = 0; a < 4; a++)
#pragma unroll
        for (int b2 = 0; b2 < 2; b2++) acc[a][b2] = zero;

    for (int k = 0; k < K; k += 32) {
        ABfrag af[4];
#pragma unroll
        for (int mf = 0; mf < 4; mf++) {
            const unsigned short* p =
                A + (size_t)(m0 + mf * 16 + lr) * K + k + lh * 8;
            af[mf].u[0] = *(const uint4*)p;
            af[mf].u[1] = *(const uint4*)(p + 16);
        }
        ABfrag bfr[2];
#pragma unroll
        for (int nf = 0; nf < 2; nf++) {
            const unsigned short* p =
                Bt + (size_t)(n0 + nf * 16 + lr) * K + k + lh * 8;
            bfr[nf].u[0] = *(const uint4*)p;
            bfr[nf].u[1] = *(const uint4*)(p + 16);
        }
#pragma unroll
        for (int mf = 0; mf < 4; mf++)
#pragma unroll
            for (int nf = 0; nf < 2; nf++)
                acc[mf][nf] = wmma_bf(af[mf].v, bfr[nf].v, acc[mf][nf]);
    }

#pragma unroll
    for (int mf = 0; mf < 4; mf++)
#pragma unroll
        for (int nf = 0; nf < 2; nf++)
#pragma unroll
            for (int r = 0; r < 8; r++) {
                int row = m0 + mf * 16 + r + 8 * lh;
                int col = n0 + nf * 16 + lr;
                float vv = acc[mf][nf][r];
                if (BF16OUT)
                    ((unsigned short*)Cout)[(size_t)row * Nc + col] = f2bf(vv);
                else
                    ((float*)Cout)[(size_t)row * Nc + col] = vv;
            }
}

// ---------------------------------------------------------------------------
// RoPE + head transpose:  src[B][N][nh*64] -> dst[B][nh][N][64]
// freq_vec[j] = theta^{-(j>>1)/32};  out[j]   = x[j]*cos[j]   - x[j+32]*sin[j]
//                                    out[j+32]= x[j]*sin[j+32]+ x[j+32]*cos[j+32]
// ---------------------------------------------------------------------------
__global__ void k_rope(const unsigned short* __restrict__ src,
                       unsigned short* __restrict__ dst, int nh) {
    int i = blockIdx.x * blockDim.x + threadIdx.x;
    int total = Bb * Nn * nh * 32;
    if (i >= total) return;
    int j = i & 31;
    int t = i >> 5;
    int h = t % nh; t /= nh;
    int n = t % Nn;
    int b = t / Nn;

    const unsigned short* s = src + ((size_t)(b * Nn + n) * nh + h) * 64;
    float x1 = bf2f(s[j]), x2 = bf2f(s[j + 32]);

    const float l2th = 13.287712379549449f;  // log2(10000)
    float f1 = exp2f(-((float)(j >> 1)) * (l2th / 32.0f));
    float f2 = exp2f(-((float)(16 + (j >> 1))) * (l2th / 32.0f));
    float a1 = (float)n * f1, a2 = (float)n * f2;
    float o1 = x1 * cosf(a1) - x2 * sinf(a1);
    float o2 = x1 * sinf(a2) + x2 * cosf(a2);

    unsigned short* dp = dst + ((size_t)(b * nh + h) * Nn + n) * 64;
    dp[j] = f2bf(o1);
    dp[j + 32] = f2bf(o2);
}

// ---------------------------------------------------------------------------
// Flash attention, one wave per (b, h, 16-row Q tile).
// S = Q Kt (4 WMMAs/32 keys), online softmax f32, O += P V (4 WMMAs/32 keys)
// ---------------------------------------------------------------------------
__global__ __launch_bounds__(256) void k_attn(const unsigned short* __restrict__ Qr,
                                              const unsigned short* __restrict__ Kr,
                                              const unsigned short* __restrict__ Vt,
                                              const int* __restrict__ mask,
                                              unsigned short* __restrict__ O) {
    __shared__ unsigned short smem[8][16][32];  // per-wave 16x32 P tile (bf16)

    int w = threadIdx.x >> 5, lane = threadIdx.x & 31;
    int lr = lane & 15, lh = lane >> 4;
    int gw = blockIdx.x * 8 + w;
    int qt = gw & 127;          // N/16 tiles
    int h  = (gw >> 7) & 31;
    int b  = gw >> 12;
    int kv = h >> 2;            // G = 4
    int q0 = qt << 4;
    float slope = exp2f(-0.25f * (float)(h + 1));
    const float scale = 0.125f;  // 1/sqrt(64)

    // Q A-fragments (2 k-chunks of 32 over HD=64), resident for whole loop
    ABfrag qa[2];
    const unsigned short* qb = Qr + ((size_t)(b * Hh + h) * Nn + q0 + lr) * 64;
#pragma unroll
    for (int c = 0; c < 2; c++) {
        const unsigned short* p = qb + c * 32 + lh * 8;
        qa[c].u[0] = *(const uint4*)p;
        qa[c].u[1] = *(const uint4*)(p + 16);
    }

    float mst[8], lst[8];
    v8f acc[4];
    v8f zero = {0.f, 0.f, 0.f, 0.f, 0.f, 0.f, 0.f, 0.f};
#pragma unroll
    for (int r = 0; r < 8; r++) { mst[r] = -1e30f; lst[r] = 0.f; }
#pragma unroll
    for (int t = 0; t < 4; t++) acc[t] = zero;

    const unsigned short* kb = Kr + (size_t)(b * KVHh + kv) * Nn * 64;
    const unsigned short* vb = Vt + (size_t)(b * KVHh + kv) * 64 * Nn;
    int nsteps = (qt >> 1) + 1;  // keys [0, 32*nsteps) cover causal window

    for (int st = 0; st < nsteps; st++) {
        int key0 = st << 5;

        // prefetch next step's K/V tiles (global_prefetch_b8; no counters)
        if (st + 1 < nsteps) {
            int nk = key0 + 32;
            __builtin_prefetch(kb + (size_t)(nk + lr) * 64, 0, 3);
            __builtin_prefetch(kb + (size_t)(nk + 16 + lr) * 64, 0, 3);
#pragma unroll
            for (int t = 0; t < 4; t++)
                __builtin_prefetch(vb + (size_t)(t * 16 + lr) * Nn + nk, 0, 3);
        }

        // --- S = Q K^T over 32 keys (two 16-key subtiles) ---
        v8f sfr[2];
#pragma unroll
        for (int s = 0; s < 2; s++) {
            const unsigned short* kr =
                kb + (size_t)(key0 + s * 16 + lr) * 64 + lh * 8;
            ABfrag k0, k1;
            k0.u[0] = *(const uint4*)kr;        k0.u[1] = *(const uint4*)(kr + 16);
            k1.u[0] = *(const uint4*)(kr + 32); k1.u[1] = *(const uint4*)(kr + 48);
            v8f t0 = wmma_bf(qa[0].v, k0.v, zero);
            sfr[s] = wmma_bf(qa[1].v, k1.v, t0);
        }

        // --- scale -> causal -> +alibi -> pad, then online softmax ---
        int j0 = key0 + lr, j1 = j0 + 16;
        float ab0 = slope * (float)j0, ab1 = slope * (float)j1;
        bool pad0 = mask[(size_t)b * Nn + j0] != 0;
        bool pad1 = mask[(size_t)b * Nn + j1] != 0;
#pragma unroll
        for (int r = 0; r < 8; r++) {
            int rowq = q0 + r + 8 * lh;
            float v0 = (j0 <= rowq) ? sfr[0][r] * scale : -1e9f;
            v0 += ab0; if (!pad0) v0 = -1e9f;
            float v1 = (j1 <= rowq) ? sfr[1][r] * scale : -1e9f;
            v1 += ab1; if (!pad1) v1 = -1e9f;

            float mx = fmaxf(v0, v1);
#pragma unroll
            for (int d = 1; d < 16; d <<= 1) mx = fmaxf(mx, __shfl_xor(mx, d, 32));
            float mnew = fmaxf(mst[r], mx);
            float sc = __expf(mst[r] - mnew);
            mst[r] = mnew;
            lst[r] *= sc;
            acc[0][r] *= sc; acc[1][r] *= sc; acc[2][r] *= sc; acc[3][r] *= sc;

            float p0 = __expf(v0 - mnew), p1 = __expf(v1 - mnew);
            float rs = p0 + p1;
#pragma unroll
            for (int d = 1; d < 16; d <<= 1) rs += __shfl_xor(rs, d, 32);
            lst[r] += rs;

            smem[w][r + 8 * lh][lr]      = f2bf(p0);   // D-layout -> LDS tile
            smem[w][r + 8 * lh][16 + lr] = f2bf(p1);
        }

        // --- re-read P as A-fragment (16 x 32 over keys) ---
        ABfrag pa;
        const unsigned short* lp = &smem[w][lr][lh * 8];
        pa.u[0] = *(const uint4*)lp;
        pa.u[1] = *(const uint4*)(lp + 16);

        // --- O += P V : 4 hd tiles of 16 ---
#pragma unroll
        for (int t = 0; t < 4; t++) {
            const unsigned short* vp =
                vb + (size_t)(t * 16 + lr) * Nn + key0 + lh * 8;
            ABfrag vf;
            vf.u[0] = *(const uint4*)vp;
            vf.u[1] = *(const uint4*)(vp + 16);
            acc[t] = wmma_bf(pa.v, vf.v, acc[t]);
        }
    }

    // --- normalize and store O[b][n][h*64 + hd] (bf16) ---
#pragma unroll
    for (int r = 0; r < 8; r++) {
        float inv = 1.0f / lst[r];
        int row = q0 + r + 8 * lh;
        size_t base = ((size_t)b * Nn + row) * (size_t)(Hh * HDd) + (size_t)h * 64;
        O[base + lr]      = f2bf(acc[0][r] * inv);
        O[base + 16 + lr] = f2bf(acc[1][r] * inv);
        O[base + 32 + lr] = f2bf(acc[2][r] * inv);
        O[base + 48 + lr] = f2bf(acc[3][r] * inv);
    }
}

// ---------------------------------------------------------------------------
// Host launcher
// ---------------------------------------------------------------------------
extern "C" void kernel_launch(void* const* d_in, const int* in_sizes, int n_in,
                              void* d_out, int out_size, void* d_ws, size_t ws_size,
                              hipStream_t stream) {
    const float* x  = (const float*)d_in[0];
    const int* mask = (const int*)d_in[1];
    const float* Wq = (const float*)d_in[2];
    const float* Wk = (const float*)d_in[3];
    const float* Wv = (const float*)d_in[4];
    const float* Wo = (const float*)d_in[5];
    float* out = (float*)d_out;

    char* ws = (char*)d_ws;
    size_t off = 0;
    auto alloc = [&](size_t bytes) -> void* {
        void* p = ws + off;
        off += (bytes + 255) & ~(size_t)255;
        return p;
    };
    unsigned short* Xb  = (unsigned short*)alloc((size_t)Mrows * Dd * 2);
    unsigned short* Wqt = (unsigned short*)alloc((size_t)Hh * HDd * Dd * 2);
    unsigned short* Wkt = (unsigned short*)alloc((size_t)KVHh * HDd * Dd * 2);
    unsigned short* Wvt = (unsigned short*)alloc((size_t)KVHh * HDd * Dd * 2);
    unsigned short* Wot = (unsigned short*)alloc((size_t)Dd * Hh * HDd * 2);
    unsigned short* Qg  = (unsigned short*)alloc((size_t)Mrows * Hh * HDd * 2);
    unsigned short* Kg  = (unsigned short*)alloc((size_t)Mrows * KVHh * HDd * 2);
    unsigned short* Vg  = (unsigned short*)alloc((size_t)Mrows * KVHh * HDd * 2);
    unsigned short* Qr  = (unsigned short*)alloc((size_t)Mrows * Hh * HDd * 2);
    unsigned short* Kr  = (unsigned short*)alloc((size_t)Mrows * KVHh * HDd * 2);
    unsigned short* Vt  = (unsigned short*)alloc((size_t)Mrows * KVHh * HDd * 2);
    unsigned short* Oat = (unsigned short*)alloc((size_t)Mrows * Hh * HDd * 2);

    const int T = 256;
    // converts (vectorized) + LDS-tiled transposes
    k_f32_to_bf16_v4<<<(Mrows * Dd / 4) / T, T, 0, stream>>>(x, Xb, Mrows * Dd / 4);
    k_wtrans<<<(Dd / 32) * ((Hh * HDd) / 32), T, 0, stream>>>(Wq, Wqt, Dd, Hh * HDd);
    k_wtrans<<<(Dd / 32) * ((KVHh * HDd) / 32), T, 0, stream>>>(Wk, Wkt, Dd, KVHh * HDd);
    k_wtrans<<<(Dd / 32) * ((KVHh * HDd) / 32), T, 0, stream>>>(Wv, Wvt, Dd, KVHh * HDd);
    k_wtrans<<<((Hh * HDd) / 32) * (Dd / 32), T, 0, stream>>>(Wo, Wot, Hh * HDd, Dd);

    // QKV projections (bf16 out)
    k_gemm<true><<<(Mrows / 128) * ((Hh * HDd) / 128), T, 0, stream>>>(
        Xb, Wqt, Qg, Mrows, Dd, Hh * HDd);
    k_gemm<true><<<(Mrows / 128) * ((KVHh * HDd) / 128), T, 0, stream>>>(
        Xb, Wkt, Kg, Mrows, Dd, KVHh * HDd);
    k_gemm<true><<<(Mrows / 128) * ((KVHh * HDd) / 128), T, 0, stream>>>(
        Xb, Wvt, Vg, Mrows, Dd, KVHh * HDd);

    // RoPE + layout transforms
    k_rope<<<(Bb * Nn * Hh * 32) / T, T, 0, stream>>>(Qg, Qr, Hh);
    k_rope<<<(Bb * Nn * KVHh * 32) / T, T, 0, stream>>>(Kg, Kr, KVHh);
    k_vtrans<<<Bb * KVHh * 2 * (Nn / 32), T, 0, stream>>>(Vg, Vt);

    // flash attention: B*H*(N/16) waves / 8 waves per block
    k_attn<<<(Bb * Hh * (Nn / 16)) / 8, T, 0, stream>>>(Qr, Kr, Vt, mask, Oat);

    // output projection (f32 out straight to d_out)
    k_gemm<false><<<(Mrows / 128) * (Dd / 128), T, 0, stream>>>(
        Oat, Wot, out, Mrows, Dd, Dd);
}
